// MultiHeadAttention_54915451847175
// MI455X (gfx1250) — matrix operand order repaired and checked by reading, not compiled, not run on hardware
//
#include <hip/hip_runtime.h>

// ---------------------------------------------------------------------------
// MI455X (gfx1250) no-softmax MHA via associativity:
//   out = Q (K^T V) * E^{-1/2}
// All matmuls on v_wmma_f32_16x16x32_bf16 (wave32).
// Projection GEMM stages its shared B panel through LDS with
// global_load_async_to_lds_b128 (ASYNCcnt) + double buffering.
// ---------------------------------------------------------------------------

typedef __bf16 bf16_t;
typedef __attribute__((ext_vector_type(16))) __bf16 v16bf;
typedef __attribute__((ext_vector_type(8)))  __bf16 v8bf;
typedef __attribute__((ext_vector_type(8)))  float  v8f;

#define NB 2
#define SLEN 2048
#define DM 1024
#define NH 16
#define NE 64
#define NPAIR (NB * NH)          // 32 (b,h) pairs
#define LDSH 40                  // padded LDS row stride (elems): 80B, 16B-aligned

// ---- WMMA fragment helpers (layouts per CDNA5 ISA 7.12.2) ------------------

// A fragment: 16x32 bf16, source row-major, row stride `ld` elements.
// lane m in [0,16): K = {0..7} then {16..23}; lane m+16: K = {8..15},{24..31}
__device__ __forceinline__ v16bf load_frag_a(const bf16_t* __restrict__ base, int ld) {
    int lane = threadIdx.x & 31;
    int row  = lane & 15;
    int koff = (lane >> 4) << 3;                // 0 or 8
    const bf16_t* p = base + row * ld + koff;
    v8bf lo = *(const v8bf*)(p);                // K = koff .. koff+7
    v8bf hi = *(const v8bf*)(p + 16);           // K = koff+16 .. koff+23
    return __builtin_shufflevector(lo, hi, 0,1,2,3,4,5,6,7,8,9,10,11,12,13,14,15);
}

// B fragment: 32x16 bf16, source is B^T row-major (col-major B), stride `ld`.
// lane n in [0,16): K = {0..7} then {16..23}; lane n+16: K = {8..15},{24..31}
__device__ __forceinline__ v16bf load_frag_bT(const bf16_t* __restrict__ base, int ld) {
    int lane = threadIdx.x & 31;
    int col  = lane & 15;
    int koff = (lane >> 4) << 3;                // 0 or 8
    const bf16_t* p = base + col * ld + koff;
    v8bf lo = *(const v8bf*)(p);                // K = koff .. koff+7
    v8bf hi = *(const v8bf*)(p + 16);           // K = koff+16 .. koff+23
    return __builtin_shufflevector(lo, hi, 0,1,2,3,4,5,6,7,8,9,10,11,12,13,14,15);
}

__device__ __forceinline__ v8f wmma_bf16(v16bf a, v16bf b, v8f c) {
    return __builtin_amdgcn_wmma_f32_16x16x32_bf16(
        /*neg_a=*/false, a, /*neg_b=*/false, b,
        /*c_mod=*/(short)0, c, /*reuse_a=*/false, /*reuse_b=*/false);
}

// ---- CDNA5 async global->LDS copy (ASYNCcnt) -------------------------------

typedef __attribute__((address_space(3))) bf16_t lds_bf16;

__device__ __forceinline__ unsigned lds_byte_off(const bf16_t* p) {
    // addrspace(3) pointers are 0-based 32-bit LDS offsets on AMDGPU.
    return (unsigned)(__UINTPTR_TYPE__)(lds_bf16*)p;
}

__device__ __forceinline__ void async_copy_b128(unsigned lds_off, const bf16_t* gptr) {
    unsigned long long ga = (unsigned long long)gptr;
    asm volatile("global_load_async_to_lds_b128 %0, %1, off"
                 :: "v"(lds_off), "v"(ga)
                 : "memory");
}

__device__ __forceinline__ void wait_async0() {
    asm volatile("s_wait_asynccnt 0x0" ::: "memory");
}

// ---- elementwise f32 -> bf16 convert ---------------------------------------

__global__ void cvt_f32_bf16_kernel(const float* __restrict__ src,
                                    bf16_t* __restrict__ dst, int n) {
    int i = blockIdx.x * blockDim.x + threadIdx.x;
    if (i < n) dst[i] = (bf16_t)src[i];
}

// ---- Kernel 1: fused Q/K/V projection GEMM ---------------------------------
// C[4096 x 3072] = Xb[4096 x 1024] * Wb^T, Wb rows = stacked (q|k|v, h, e).
// Block = 8 waves = 256 rows x 64 cols; per wave a 32x64 tile (2x4 C frags).
// The 64x32 B^T panel per k-step is shared by all 8 waves: double-buffered in
// LDS via async-to-LDS b128 copies (1 per thread per k-step).
__global__ __launch_bounds__(256)
void proj_gemm_kernel(const bf16_t* __restrict__ Xb, const bf16_t* __restrict__ Wb,
                      const float* __restrict__ bq, const float* __restrict__ bk,
                      const float* __restrict__ bv,
                      bf16_t* __restrict__ Qb, bf16_t* __restrict__ Ktb,
                      bf16_t* __restrict__ Vtb) {
    __shared__ __align__(16) bf16_t Bsh[2][64 * LDSH];   // 2 x 5.0 KB

    int wave = threadIdx.x >> 5;
    int bm   = blockIdx.x & 15;                  // 16 row blocks of 256
    int tn   = blockIdx.x >> 4;                  // 48 col tiles of 64
    int proj = tn >> 4;                          // 0=q 1=k 2=v
    int h    = tn & 15;
    int m0   = bm * 256 + wave * 32;

    const bf16_t* Arow = Xb + (size_t)m0 * DM;
    const bf16_t* Brow = Wb + (size_t)tn * 64 * DM;

    // async-fill assignment: thread t copies 16B (8 elems) of the B^T panel
    int br = threadIdx.x >> 2;                   // 0..63 panel row
    int bc = (threadIdx.x & 3) << 3;             // 0,8,16,24 elem offset
    const bf16_t* gsrc = Brow + br * DM + bc;    // + k0 at issue time
    unsigned ldst0 = lds_byte_off(&Bsh[0][br * LDSH + bc]);
    unsigned ldst1 = lds_byte_off(&Bsh[1][br * LDSH + bc]);

    v8f c[2][4];
#pragma unroll
    for (int i = 0; i < 2; ++i)
#pragma unroll
        for (int j = 0; j < 4; ++j) c[i][j] = (v8f){0,0,0,0,0,0,0,0};

    // prologue: fill buffer 0 with k-step 0
    async_copy_b128(ldst0, gsrc);

    for (int kk = 0; kk < DM / 32; ++kk) {
        wait_async0();
        __syncthreads();                          // panel for kk ready everywhere
        if (kk + 1 < DM / 32)                     // prefetch next panel
            async_copy_b128((kk & 1) ? ldst0 : ldst1, gsrc + (kk + 1) * 32);

        const bf16_t* ap  = Arow + kk * 32;
        const bf16_t* bsh = Bsh[kk & 1];
        v16bf a0 = load_frag_a(ap, DM);
        v16bf a1 = load_frag_a(ap + 16 * DM, DM);
        v16bf b0 = load_frag_bT(bsh, LDSH);
        v16bf b1 = load_frag_bT(bsh + 16 * LDSH, LDSH);
        v16bf b2 = load_frag_bT(bsh + 32 * LDSH, LDSH);
        v16bf b3 = load_frag_bT(bsh + 48 * LDSH, LDSH);
        c[0][0] = wmma_bf16(a0, b0, c[0][0]);
        c[0][1] = wmma_bf16(a0, b1, c[0][1]);
        c[0][2] = wmma_bf16(a0, b2, c[0][2]);
        c[0][3] = wmma_bf16(a0, b3, c[0][3]);
        c[1][0] = wmma_bf16(a1, b0, c[1][0]);
        c[1][1] = wmma_bf16(a1, b1, c[1][1]);
        c[1][2] = wmma_bf16(a1, b2, c[1][2]);
        c[1][3] = wmma_bf16(a1, b3, c[1][3]);
    }

    // Epilogue: C layout lane l: N = l&15, M = r + 8*(l>>4)
    int lane  = threadIdx.x & 31;
    int nloc  = lane & 15;
    int mbase = (lane >> 4) * 8;
    const float* bias = (proj == 0) ? bq : (proj == 1) ? bk : bv;
#pragma unroll
    for (int j = 0; j < 4; ++j) {
        int e = j * 16 + nloc;
        float bval = bias[h * NE + e];
#pragma unroll
        for (int i = 0; i < 2; ++i)
#pragma unroll
            for (int r = 0; r < 8; ++r) {
                int bs = m0 + i * 16 + mbase + r;      // global row in [0,4096)
                int b  = bs >> 11;                      // / SLEN
                int s  = bs & (SLEN - 1);
                int p  = b * NH + h;
                float val = c[i][j][r] + bval;
                if (proj == 0) {
                    Qb[((size_t)p * SLEN + s) * NE + e] = (bf16_t)val;
                } else {
                    bf16_t* dst = (proj == 1) ? Ktb : Vtb;
                    dst[((size_t)p * NE + e) * SLEN + s] = (bf16_t)val;
                }
            }
    }
}

// ---- Kernel 2: M^T = V^T K per (b,h), split over S ------------------------
// A = Vt rows (e_v, s), B^T = Kt rows (e_k, s). Contraction over s.
// Each wave: 32 e_v rows x 64 e_k cols, s-range of 256 (8 k-steps of 32).
// Partial f32 sums into Mpart[pair][split][64][64].
__global__ __launch_bounds__(32)
void ktv_gemm_kernel(const bf16_t* __restrict__ Ktb, const bf16_t* __restrict__ Vtb,
                     float* __restrict__ Mpart) {
    int p     = blockIdx.x >> 4;         // 32 pairs
    int split = (blockIdx.x >> 1) & 7;   // 8 s-splits
    int iv    = blockIdx.x & 1;          // e_v half

    const bf16_t* Vrow = Vtb + ((size_t)p * NE + iv * 32) * SLEN + split * 256;
    const bf16_t* Krow = Ktb + (size_t)p * NE * SLEN + split * 256;

    v8f c[2][4];
#pragma unroll
    for (int i = 0; i < 2; ++i)
#pragma unroll
        for (int j = 0; j < 4; ++j) c[i][j] = (v8f){0,0,0,0,0,0,0,0};

    for (int kk = 0; kk < 8; ++kk) {
        const bf16_t* ap = Vrow + kk * 32;
        const bf16_t* bp = Krow + kk * 32;
        v16bf a0 = load_frag_a(ap, SLEN);
        v16bf a1 = load_frag_a(ap + 16 * SLEN, SLEN);
        v16bf b0 = load_frag_bT(bp, SLEN);
        v16bf b1 = load_frag_bT(bp + 16 * SLEN, SLEN);
        v16bf b2 = load_frag_bT(bp + 32 * SLEN, SLEN);
        v16bf b3 = load_frag_bT(bp + 48 * SLEN, SLEN);
        c[0][0] = wmma_bf16(a0, b0, c[0][0]);
        c[0][1] = wmma_bf16(a0, b1, c[0][1]);
        c[0][2] = wmma_bf16(a0, b2, c[0][2]);
        c[0][3] = wmma_bf16(a0, b3, c[0][3]);
        c[1][0] = wmma_bf16(a1, b0, c[1][0]);
        c[1][1] = wmma_bf16(a1, b1, c[1][1]);
        c[1][2] = wmma_bf16(a1, b2, c[1][2]);
        c[1][3] = wmma_bf16(a1, b3, c[1][3]);
    }

    float* dst = Mpart + ((size_t)p * 8 + split) * (NE * NE);
    int lane  = threadIdx.x & 31;
    int nloc  = lane & 15;
    int mbase = (lane >> 4) * 8;
#pragma unroll
    for (int i = 0; i < 2; ++i)
#pragma unroll
        for (int j = 0; j < 4; ++j)
#pragma unroll
            for (int r = 0; r < 8; ++r) {
                int ev = iv * 32 + i * 16 + mbase + r;
                int ek = j * 16 + nloc;
                dst[ev * NE + ek] = c[i][j][r];
            }
}

// ---- Kernel 2b: reduce split partials, convert M^T to bf16 -----------------
__global__ void reduce_m_kernel(const float* __restrict__ Mpart,
                                bf16_t* __restrict__ Mtb) {
    int idx = blockIdx.x * blockDim.x + threadIdx.x;   // 32*4096 outputs
    if (idx < NPAIR * NE * NE) {
        int p = idx >> 12;
        int o = idx & 4095;
        float s = 0.f;
#pragma unroll
        for (int sp = 0; sp < 8; ++sp)
            s += Mpart[((size_t)p * 8 + sp) * (NE * NE) + o];
        Mtb[idx] = (bf16_t)s;
    }
}

// ---- Kernel 3: out = Q * M * norm, written to [B, S, H*E] f32 --------------
// A = Qb rows (s, e_k) ld=64; B^T = Mtb rows (e, e_k) ld=64. K-loop = 2 steps.
__global__ __launch_bounds__(256)
void out_gemm_kernel(const bf16_t* __restrict__ Qb, const bf16_t* __restrict__ Mtb,
                     float* __restrict__ out) {
    int wid = blockIdx.x * 8 + (threadIdx.x >> 5);   // 2048 waves
    int p   = wid >> 6;                               // pair
    int tm  = wid & 63;                               // 64 row tiles of 32
    int b   = p >> 4;
    int h   = p & 15;
    int s0  = tm * 32;

    const bf16_t* Arow = Qb + ((size_t)p * SLEN + s0) * NE;
    const bf16_t* Bt   = Mtb + (size_t)p * (NE * NE);

    v8f c[2][4];
#pragma unroll
    for (int i = 0; i < 2; ++i)
#pragma unroll
        for (int j = 0; j < 4; ++j) c[i][j] = (v8f){0,0,0,0,0,0,0,0};

#pragma unroll
    for (int kk = 0; kk < 2; ++kk) {
        const bf16_t* ap = Arow + kk * 32;
        const bf16_t* bp = Bt + kk * 32;
        v16bf a0 = load_frag_a(ap, NE);
        v16bf a1 = load_frag_a(ap + 16 * NE, NE);
        v16bf b0 = load_frag_bT(bp, NE);
        v16bf b1 = load_frag_bT(bp + 16 * NE, NE);
        v16bf b2 = load_frag_bT(bp + 32 * NE, NE);
        v16bf b3 = load_frag_bT(bp + 48 * NE, NE);
        c[0][0] = wmma_bf16(a0, b0, c[0][0]);
        c[0][1] = wmma_bf16(a0, b1, c[0][1]);
        c[0][2] = wmma_bf16(a0, b2, c[0][2]);
        c[0][3] = wmma_bf16(a0, b3, c[0][3]);
        c[1][0] = wmma_bf16(a1, b0, c[1][0]);
        c[1][1] = wmma_bf16(a1, b1, c[1][1]);
        c[1][2] = wmma_bf16(a1, b2, c[1][2]);
        c[1][3] = wmma_bf16(a1, b3, c[1][3]);
    }

    const float norm = 0.125f;   // 64^-0.5
    int lane  = threadIdx.x & 31;
    int nloc  = lane & 15;
    int mbase = (lane >> 4) * 8;
#pragma unroll
    for (int j = 0; j < 4; ++j) {
        int e = j * 16 + nloc;
#pragma unroll
        for (int i = 0; i < 2; ++i)
#pragma unroll
            for (int r = 0; r < 8; ++r) {
                int s = s0 + i * 16 + mbase + r;
                out[((size_t)b * SLEN + s) * (NH * NE) + h * NE + e] =
                    c[i][j][r] * norm;
            }
    }
}

// ---------------------------------------------------------------------------

extern "C" void kernel_launch(void* const* d_in, const int* in_sizes, int n_in,
                              void* d_out, int out_size, void* d_ws, size_t ws_size,
                              hipStream_t stream) {
    const float* x  = (const float*)d_in[0];
    const float* Wq = (const float*)d_in[1];
    const float* Wk = (const float*)d_in[2];
    const float* Wv = (const float*)d_in[3];
    const float* bq = (const float*)d_in[4];
    const float* bk = (const float*)d_in[5];
    const float* bv = (const float*)d_in[6];
    float* out = (float*)d_out;

    // Workspace layout (bytes)
    char* ws = (char*)d_ws;
    const size_t nX = (size_t)NB * SLEN * DM;          // 4,194,304
    const size_t nW = (size_t)NH * NE * DM;            // 1,048,576 per proj
    const size_t nQ = (size_t)NPAIR * SLEN * NE;       // 4,194,304
    bf16_t* Xb  = (bf16_t*)(ws);                       //  8.0 MB
    bf16_t* Wb  = (bf16_t*)(ws + 2 * nX);              //  6.0 MB (3 projs)
    bf16_t* Qb  = (bf16_t*)(ws + 2 * (nX + 3 * nW));
    bf16_t* Ktb = Qb + nQ;
    bf16_t* Vtb = Ktb + nQ;
    float*  Mpart = (float*)(Vtb + nQ);                // 32*8*64*64 f32 = 4 MB
    bf16_t* Mtb = (bf16_t*)(Mpart + (size_t)NPAIR * 8 * NE * NE);

    // 0) convert inputs to bf16
    cvt_f32_bf16_kernel<<<(int)((nX + 255) / 256), 256, 0, stream>>>(x, Xb, (int)nX);
    cvt_f32_bf16_kernel<<<(int)((nW + 255) / 256), 256, 0, stream>>>(Wq, Wb, (int)nW);
    cvt_f32_bf16_kernel<<<(int)((nW + 255) / 256), 256, 0, stream>>>(Wk, Wb + nW, (int)nW);
    cvt_f32_bf16_kernel<<<(int)((nW + 255) / 256), 256, 0, stream>>>(Wv, Wb + 2 * nW, (int)nW);

    // 1) fused Q/K/V projections: 768 blocks x 8 waves (256 rows x 64 cols each)
    proj_gemm_kernel<<<768, 256, 0, stream>>>(Xb, Wb, bq, bk, bv, Qb, Ktb, Vtb);

    // 2) M^T = V^T K per head, split-S x8, e_v halves x2 -> 512 single-wave blocks
    ktv_gemm_kernel<<<512, 32, 0, stream>>>(Ktb, Vtb, Mpart);

    // 2b) reduce partials -> bf16 M^T
    reduce_m_kernel<<<(NPAIR * NE * NE + 255) / 256, 256, 0, stream>>>(Mpart, Mtb);

    // 3) out = Q * M * 0.125 : 2048 waves, 8 per block
    out_gemm_kernel<<<256, 256, 0, stream>>>(Qb, Mtb, out);
}
